// AttentionModel_18313740550177
// MI455X (gfx1250) — matrix operand order repaired and checked by reading, not compiled, not run on hardware
//
#include <hip/hip_runtime.h>
#include <hip/hip_bf16.h>

// ---------------------------------------------------------------------------
// Problem constants (from the reference): x[8,2048,512] fp32, 4x [512,512] W.
// ---------------------------------------------------------------------------
constexpr int Bb = 8;
constexpr int Ss = 2048;
constexpr int Dd = 512;
constexpr int Mrows = Bb * Ss;          // 16384

typedef __attribute__((ext_vector_type(16))) __bf16 v16bf;
typedef __attribute__((ext_vector_type(8)))  __bf16 v8bf;
typedef __attribute__((ext_vector_type(8)))  float  v8f;

// TDM descriptor word groups (per CDNA5_HIP.md probe: ROCm7.2 = 5 args)
typedef __attribute__((ext_vector_type(4))) unsigned int u32x4;
typedef __attribute__((ext_vector_type(8))) int          i32x8;
typedef __attribute__((ext_vector_type(4))) int          i32x4;

#if __has_builtin(__builtin_amdgcn_tensor_load_to_lds) && \
    __has_builtin(__builtin_amdgcn_s_wait_tensorcnt)
#define HAVE_TDM 1
#else
#define HAVE_TDM 0
#endif

union AFrag { v16bf v; v8bf h[2]; };

__device__ __forceinline__ __bf16 f2bf(float f) {
    union { float f; unsigned u; } a; a.f = f;
    unsigned r = a.u + 0x7FFFu + ((a.u >> 16) & 1u);   // round-to-nearest-even
    union { unsigned short s; __bf16 b; } o;
    o.s = (unsigned short)(r >> 16);
    return o.b;
}

#if HAVE_TDM
// Issue a TDM 2D tile copy Global->LDS: `rows` rows of `cols` bf16 elements,
// source row stride `srcStride` elements, packed contiguously at LDS byte
// offset `ldsOff`.  D# layout per CDNA5 ISA ch.8 (group0/group1 bitfields).
__device__ __forceinline__ void tdm_load_2d_bf16(const void* gsrc, unsigned ldsOff,
                                                 unsigned cols, unsigned rows,
                                                 unsigned srcStride) {
    unsigned long long ga = (unsigned long long)gsrc;
    u32x4 g0;
    g0.x = 1u;                                            // count=1, user D#
    g0.y = ldsOff;                                        // lds_addr (bytes)
    g0.z = (unsigned)ga;                                  // global_addr[31:0]
    g0.w = (unsigned)((ga >> 32) & 0x01FFFFFFu)           // global_addr[56:32]
         | 0x80000000u;                                   // type=2 ("image")
    i32x8 g1;
    g1[0] = 0x00010000;                                   // data_size=1 (2 bytes)
    g1[1] = (int)(cols << 16);                            // tensor_dim0[15:0]
    g1[2] = (int)((cols >> 16) | (rows << 16));           // dim0[31:16] | dim1[15:0]
    g1[3] = (int)((rows >> 16) | (cols << 16));           // dim1[31:16] | tile_dim0
    g1[4] = (int)rows;                                    // tile_dim1 (tile_dim2=0)
    g1[5] = (int)srcStride;                               // tensor_dim0_stride[31:0]
    g1[6] = 0;                                            // stride[47:32] | dim1_stride lo
    g1[7] = 0;
    i32x4 z4 = {0, 0, 0, 0};
#if __clang_major__ >= 23
    i32x8 z8 = {0, 0, 0, 0, 0, 0, 0, 0};
    __builtin_amdgcn_tensor_load_to_lds(g0, g1, z4, z4, z8, 0);
#else
    __builtin_amdgcn_tensor_load_to_lds(g0, g1, z4, z4, 0);
#endif
}
#endif

// ---------------------------------------------------------------------------
// fp32 -> bf16 conversion
// ---------------------------------------------------------------------------
__global__ void cvt_f32_bf16(const float* __restrict__ s, __bf16* __restrict__ d, int n) {
    int i = blockIdx.x * 256 + threadIdx.x;
    if (i < n) d[i] = f2bf(s[i]);
}

// ---------------------------------------------------------------------------
// C[M,N] = A[M,K] @ W[N,K]^T + bias  (torch Linear). bf16 inputs, fp32 acc.
// Block: 128 threads = 4 waves, each wave owns 16 rows x 64 cols.
// A-frag layout: lane holds row lane%16, K = (lane/16)*16 .. +15 contiguous
// (matches B-frag k order).
// B-frag: lane holds col lane%16, K = (lane/16)*16 .. +15 contiguous.
// ---------------------------------------------------------------------------
template<bool F32OUT>
__global__ __launch_bounds__(128)
void gemm_bias(const __bf16* __restrict__ A, const __bf16* __restrict__ W,
               const float* __restrict__ bias, void* __restrict__ Cout,
               int M, int N, int K) {
    const int lane = threadIdx.x & 31;
    const int wave = threadIdx.x >> 5;
    const int half = lane >> 4;
    const int l16  = lane & 15;
    const int col0 = blockIdx.x * 64;
    const int row0 = blockIdx.y * 64 + wave * 16;

    v8f acc[4] = {};
    for (int k0 = 0; k0 < K; k0 += 32) {
        AFrag a;
        const __bf16* ap = A + (size_t)(row0 + l16) * K + k0 + half * 16;
        a.h[0] = *(const v8bf*)(ap);
        a.h[1] = *(const v8bf*)(ap + 8);
        if (k0 + 32 < K) {                       // global_prefetch next K chunk
            __builtin_prefetch((const void*)(ap + 32), 0, 1);
            __builtin_prefetch((const void*)(W + (size_t)(col0 + l16) * K + k0 + 32 + half * 16), 0, 1);
        }
        #pragma unroll
        for (int t = 0; t < 4; ++t) {
            const __bf16* bp = W + (size_t)(col0 + t * 16 + l16) * K + k0 + half * 16;
            v16bf bm = *(const v16bf*)bp;
            acc[t] = __builtin_amdgcn_wmma_f32_16x16x32_bf16(
                false, a.v, false, bm, (short)0, acc[t], false, false);
        }
    }
    #pragma unroll
    for (int t = 0; t < 4; ++t) {
        int n = col0 + t * 16 + l16;
        float bv = bias[n];
        #pragma unroll
        for (int i = 0; i < 8; ++i) {
            int m = row0 + i + half * 8;
            float v = acc[t][i] + bv;
            if (F32OUT) ((float*)Cout)[(size_t)m * N + n] = v;
            else        ((__bf16*)Cout)[(size_t)m * N + n] = f2bf(v);
        }
    }
}

// ---------------------------------------------------------------------------
// Flash attention: block = (batch, 16-query tile), 4 waves split D=512.
// Keys streamed 32 at a time: K tile staged by the Tensor Data Mover
// (Global->LDS DMA, zero VGPR traffic), V tile transposed through registers.
// Online softmax; ctx accumulators live in VGPRs.
// ---------------------------------------------------------------------------
__global__ __launch_bounds__(128)
void attn(const __bf16* __restrict__ Q, const __bf16* __restrict__ Kx,
          const __bf16* __restrict__ Vx, __bf16* __restrict__ Ctx) {
    extern __shared__ char smem[];
    __bf16* kt  = (__bf16*)smem;                  // [32][512]   32 KB (TDM dest, offset 0)
    __bf16* vT  = kt + 32 * 512;                  // [512][32]   32 KB (transposed)
    float*  sc  = (float*)(vT + 512 * 32);        // [4][16][32]  8 KB partial scores
    __bf16* pb  = (__bf16*)(sc + 4 * 16 * 32);    // [16][32]     1 KB probs (bf16)
    float*  mrow = (float*)(pb + 16 * 32);        // [16] running max
    float*  lrow = mrow + 16;                     // [16] running sum
    float*  srow = lrow + 16;                     // [16] rescale factor

    const int tid  = threadIdx.x;
    const int lane = tid & 31;
    const int wave = tid >> 5;
    const int half = lane >> 4;
    const int l16  = lane & 15;

    const int qt    = blockIdx.x;
    const int b     = qt / (Ss / 16);
    const int qrow0 = (qt % (Ss / 16)) * 16;
    const int d0    = wave * 128;                 // this wave's D slice

    // Q fragments for this wave's 128-wide D slice (persist across key loop)
    v16bf qf[4];
    #pragma unroll
    for (int kk = 0; kk < 4; ++kk) {
        AFrag u;
        const __bf16* qp = Q + ((size_t)b * Ss + qrow0 + l16) * Dd + d0 + kk * 32 + half * 16;
        u.h[0] = *(const v8bf*)qp;
        u.h[1] = *(const v8bf*)(qp + 8);
        qf[kk] = u.v;
    }

    v8f cacc[8] = {};
    if (tid < 16) { mrow[tid] = -1e30f; lrow[tid] = 0.0f; }
    __syncthreads();

    for (int key0 = 0; key0 < Ss; key0 += 32) {
#if HAVE_TDM
        // ---- K tile via Tensor Data Mover: 32x512 bf16, row stride 512 ----
        if (wave == 0) {
            tdm_load_2d_bf16(Kx + ((size_t)b * Ss + key0) * Dd,
                             /*ldsOff=*/0u, /*cols=*/(unsigned)Dd,
                             /*rows=*/32u, /*srcStride=*/(unsigned)Dd);
            __builtin_amdgcn_s_wait_tensorcnt(0);
        }
        // ---- V tile: load + transpose into LDS ----
        for (int e = tid; e < 32 * 512 / 8; e += 128) {
            int key = e >> 6;
            int dc  = (e & 63) * 8;
            v8bf vv = *(const v8bf*)(Vx + ((size_t)b * Ss + key0 + key) * Dd + dc);
            #pragma unroll
            for (int j = 0; j < 8; ++j) vT[(dc + j) * 32 + key] = vv[j];
        }
#else
        // ---- fallback: cooperative stage of K (natural) and V (transposed) ----
        for (int e = tid; e < 32 * 512 / 8; e += 128) {
            int key = e >> 6;
            int dc  = (e & 63) * 8;
            const size_t gro = ((size_t)b * Ss + key0 + key) * Dd + dc;
            v8bf kv = *(const v8bf*)(Kx + gro);
            *(v8bf*)(kt + key * 512 + dc) = kv;
            v8bf vv = *(const v8bf*)(Vx + gro);
            #pragma unroll
            for (int j = 0; j < 8; ++j) vT[(dc + j) * 32 + key] = vv[j];
        }
#endif
        __syncthreads();

        // ---- partial scores: q[16,128] @ k[32,128]^T -> 16x32 per wave ----
        v8f sacc[2] = {};
        #pragma unroll
        for (int kb = 0; kb < 2; ++kb)
            #pragma unroll
            for (int kk = 0; kk < 4; ++kk) {
                const __bf16* kp = kt + (kb * 16 + l16) * 512 + d0 + kk * 32 + half * 16;
                v16bf bm = *(const v16bf*)kp;
                sacc[kb] = __builtin_amdgcn_wmma_f32_16x16x32_bf16(
                    false, qf[kk], false, bm, (short)0, sacc[kb], false, false);
            }
        #pragma unroll
        for (int kb = 0; kb < 2; ++kb)
            #pragma unroll
            for (int i = 0; i < 8; ++i)
                sc[(wave * 16 + i + half * 8) * 32 + kb * 16 + l16] = sacc[kb][i];
        __syncthreads();

        // ---- cross-wave reduce + scale by 1/sqrt(D) ----
        for (int e = tid; e < 512; e += 128) {
            float s = sc[e] + sc[512 + e] + sc[1024 + e] + sc[1536 + e];
            sc[e] = s * 0.04419417382415922f;   // 1/sqrt(512)
        }
        __syncthreads();

        // ---- online softmax row state (16 rows, one thread each) ----
        if (tid < 16) {
            int r = tid;
            float mold = mrow[r];
            float mx = mold;
            #pragma unroll 4
            for (int c = 0; c < 32; ++c) mx = fmaxf(mx, sc[r * 32 + c]);
            float corr = __expf(mold - mx);
            float rs = 0.0f;
            #pragma unroll 4
            for (int c = 0; c < 32; ++c) {
                float p = __expf(sc[r * 32 + c] - mx);
                rs += p;
                pb[r * 32 + c] = f2bf(p);
            }
            lrow[r] = lrow[r] * corr + rs;
            mrow[r] = mx;
            srow[r] = corr;
        }
        __syncthreads();

        // ---- rescale ctx accumulators ----
        float f[8];
        #pragma unroll
        for (int i = 0; i < 8; ++i) f[i] = srow[i + half * 8];
        #pragma unroll
        for (int t = 0; t < 8; ++t)
            #pragma unroll
            for (int i = 0; i < 8; ++i) cacc[t][i] *= f[i];

        // ---- ctx += P[16,32] @ V[32,128-slice] ----
        AFrag pu;
        pu.h[0] = *(const v8bf*)(pb + l16 * 32 + half * 16);
        pu.h[1] = *(const v8bf*)(pb + l16 * 32 + half * 16 + 8);
        #pragma unroll
        for (int t = 0; t < 8; ++t) {
            const __bf16* vp = vT + (d0 + t * 16 + l16) * 32 + half * 16;
            v16bf vm = *(const v16bf*)vp;
            cacc[t] = __builtin_amdgcn_wmma_f32_16x16x32_bf16(
                false, pu.v, false, vm, (short)0, cacc[t], false, false);
        }
        __syncthreads();   // kt/vT/pb consumed; safe to restage
    }

    // ---- normalize and store ctx (bf16) ----
    float linv[8];
    #pragma unroll
    for (int i = 0; i < 8; ++i) linv[i] = 1.0f / lrow[i + half * 8];
    #pragma unroll
    for (int t = 0; t < 8; ++t)
        #pragma unroll
        for (int i = 0; i < 8; ++i) {
            int m = qrow0 + i + half * 8;
            int n = d0 + t * 16 + l16;
            Ctx[((size_t)b * Ss + m) * Dd + n] = f2bf(cacc[t][i] * linv[i]);
        }
}

// ---------------------------------------------------------------------------
// Host launcher
// ---------------------------------------------------------------------------
extern "C" void kernel_launch(void* const* d_in, const int* in_sizes, int n_in,
                              void* d_out, int out_size, void* d_ws, size_t ws_size,
                              hipStream_t stream) {
    const float* x  = (const float*)d_in[0];
    const float* Wq = (const float*)d_in[1];
    const float* bq = (const float*)d_in[2];
    const float* Wk = (const float*)d_in[3];
    const float* bk = (const float*)d_in[4];
    const float* Wv = (const float*)d_in[5];
    const float* bv = (const float*)d_in[6];
    const float* Wo = (const float*)d_in[7];
    const float* bo = (const float*)d_in[8];

    const size_t nX = (size_t)Mrows * Dd;      // 8,388,608
    const size_t nW = (size_t)Dd * Dd;         // 262,144

    char* ws = (char*)d_ws;
    size_t off = 0;
    __bf16* xb  = (__bf16*)(ws + off); off += nX * 2;
    __bf16* wqb = (__bf16*)(ws + off); off += nW * 2;
    __bf16* wkb = (__bf16*)(ws + off); off += nW * 2;
    __bf16* wvb = (__bf16*)(ws + off); off += nW * 2;
    __bf16* wob = (__bf16*)(ws + off); off += nW * 2;
    __bf16* qb  = (__bf16*)(ws + off); off += nX * 2;
    __bf16* kb_ = (__bf16*)(ws + off); off += nX * 2;
    __bf16* vb  = (__bf16*)(ws + off); off += nX * 2;
    __bf16* cb  = (__bf16*)(ws + off); off += nX * 2;   // total ~82 MB

    // 1) convert inputs to bf16
    cvt_f32_bf16<<<dim3((nX + 255) / 256), dim3(256), 0, stream>>>(x,  xb,  (int)nX);
    cvt_f32_bf16<<<dim3((nW + 255) / 256), dim3(256), 0, stream>>>(Wq, wqb, (int)nW);
    cvt_f32_bf16<<<dim3((nW + 255) / 256), dim3(256), 0, stream>>>(Wk, wkb, (int)nW);
    cvt_f32_bf16<<<dim3((nW + 255) / 256), dim3(256), 0, stream>>>(Wv, wvb, (int)nW);
    cvt_f32_bf16<<<dim3((nW + 255) / 256), dim3(256), 0, stream>>>(Wo, wob, (int)nW);

    // 2) q, k, v projections (bf16 out)
    dim3 ggrid(Dd / 64, Mrows / 64);
    gemm_bias<false><<<ggrid, dim3(128), 0, stream>>>(xb, wqb, bq, qb,  Mrows, Dd, Dd);
    gemm_bias<false><<<ggrid, dim3(128), 0, stream>>>(xb, wkb, bk, kb_, Mrows, Dd, Dd);
    gemm_bias<false><<<ggrid, dim3(128), 0, stream>>>(xb, wvb, bv, vb,  Mrows, Dd, Dd);

    // 3) flash attention
    const size_t smem = 32 * 512 * 2 + 512 * 32 * 2 + 4 * 16 * 32 * 4
                      + 16 * 32 * 2 + 3 * 16 * 4;   // ~74.9 KB
    attn<<<dim3(Bb * (Ss / 16)), dim3(128), smem, stream>>>(qb, kb_, vb, cb);

    // 4) output projection (fp32 out, straight to d_out)
    gemm_bias<true><<<ggrid, dim3(128), 0, stream>>>(cb, wob, bo, (float*)d_out, Mrows, Dd, Dd);
}
